// MixtralSparseMoeBlock_22342419874333
// MI455X (gfx1250) — matrix operand order repaired and checked by reading, not compiled, not run on hardware
//
#include <hip/hip_runtime.h>
#include <hip/hip_bf16.h>

// ---------------------------------------------------------------------------
// Mixtral sparse MoE block for MI455X (gfx1250, wave32, WMMA bf16).
//   out  = sum_{e in top2(t)} w_e(t) * ( w2[e] @ ( silu(w1[e] x) * (w3[e] x) ) )
//   plus raw router logits.
// ~360 GFLOP routed work vs ~530MB HBM traffic => compute-bound. Weights are
// pre-converted fp32->bf16 once (bandwidth pass), tokens gathered into dense
// 32-row per-expert tiles; every B fragment is reused for 2 WMMAs (two
// M-subtiles per wave) to halve L2 bandwidth per matrix op.
// ---------------------------------------------------------------------------

#define T_TOK 8192
#define HDIM  1024
#define FDIM  3584
#define NEXP  8

typedef __attribute__((ext_vector_type(16))) __bf16 v16bf;
typedef __attribute__((ext_vector_type(8)))  __bf16 v8bf;
typedef __attribute__((ext_vector_type(8)))  float  v8f;

// --- A-fragment (16x32 bf16) from LDS, row-major with given row stride.
// Layout: lanes 0-15 row=lane, K 0..15; lanes 16-31 row=lane-16,
// K 16..31 => two contiguous 16B segments per lane (matches B).
static __device__ __forceinline__ v16bf lds_load_a(const __bf16* base,
                                                   int rowStride, int lane,
                                                   int k0) {
  const int r  = lane & 15;
  const int hi = (lane >> 4) & 1;
  const __bf16* p = base + r * rowStride + k0 + hi * 16;
  union { v16bf v; v8bf h[2]; } u;
  u.h[0] = *(const v8bf*)(p);
  u.h[1] = *(const v8bf*)(p + 8);
  return u.v;
}

// --- B-fragment (32x16 bf16): lane holds column (lane&15), 16 consecutive K
// values starting at (lane>>4)*16. p points at those 16 weights.
static __device__ __forceinline__ v16bf ld_b(const float* p) {
  v16bf b;
  const float4* q = (const float4*)p;
#pragma unroll
  for (int i = 0; i < 4; ++i) {
    float4 f = q[i];
    b[4 * i + 0] = (__bf16)f.x;
    b[4 * i + 1] = (__bf16)f.y;
    b[4 * i + 2] = (__bf16)f.z;
    b[4 * i + 3] = (__bf16)f.w;
  }
  return b;
}

static __device__ __forceinline__ v16bf ld_b(const __bf16* p) {
  union { v16bf v; v8bf h[2]; } u;   // 32 contiguous bytes, two b128 loads
  u.h[0] = ((const v8bf*)p)[0];
  u.h[1] = ((const v8bf*)p)[1];
  return u.v;
}

static __device__ __forceinline__ v8f wmma_bf16(v16bf a, v16bf b, v8f c) {
  return __builtin_amdgcn_wmma_f32_16x16x32_bf16(false, a, false, b,
                                                 (short)0, c, false, false);
}

// ---------------------------------------------------------------------------
// fp32 -> bf16 weight pre-conversion (pure bandwidth, ~23us for 352MB read).
// ---------------------------------------------------------------------------
__global__ __launch_bounds__(256) void cvt_bf16_kernel(
    const float* __restrict__ src, __bf16* __restrict__ dst, size_t n) {
  size_t i = ((size_t)blockIdx.x * blockDim.x + threadIdx.x) * 8;
  if (i >= n) return;
  float4 a = ((const float4*)(src + i))[0];
  float4 b = ((const float4*)(src + i))[1];
  v8bf o;
  o[0] = (__bf16)a.x; o[1] = (__bf16)a.y; o[2] = (__bf16)a.z; o[3] = (__bf16)a.w;
  o[4] = (__bf16)b.x; o[5] = (__bf16)b.y; o[6] = (__bf16)b.z; o[7] = (__bf16)b.w;
  *(v8bf*)(dst + i) = o;
}

// ---------------------------------------------------------------------------
// Router: one wave per token. logits = x @ gate_w^T, softmax, top-2,
// normalized weights; append (token, weight) to per-expert lists.
// ---------------------------------------------------------------------------
__global__ __launch_bounds__(256) void router_kernel(
    const float* __restrict__ x, const float* __restrict__ gw,
    float* __restrict__ logits_out, int* __restrict__ cnt,
    int* __restrict__ tlist, float* __restrict__ twgt) {
  const int lane = threadIdx.x & 31;
  const int t = blockIdx.x * 8 + (threadIdx.x >> 5);

  float acc[NEXP];
#pragma unroll
  for (int e = 0; e < NEXP; ++e) acc[e] = 0.f;

  const float* xr = x + (size_t)t * HDIM;
  for (int h = lane; h < HDIM; h += 32) {
    float xv = xr[h];
#pragma unroll
    for (int e = 0; e < NEXP; ++e) acc[e] += xv * gw[e * HDIM + h];
  }
#pragma unroll
  for (int e = 0; e < NEXP; ++e) {
#pragma unroll
    for (int off = 16; off > 0; off >>= 1)
      acc[e] += __shfl_down(acc[e], off, 32);
  }

  if (lane == 0) {
    float mx = acc[0];
#pragma unroll
    for (int e = 1; e < NEXP; ++e) mx = fmaxf(mx, acc[e]);
    float p[NEXP];
#pragma unroll
    for (int e = 0; e < NEXP; ++e) {
      logits_out[t * NEXP + e] = acc[e];
      p[e] = __expf(acc[e] - mx);
    }
    int i0 = 0;
#pragma unroll
    for (int e = 1; e < NEXP; ++e)
      if (p[e] > p[i0]) i0 = e;
    int i1 = (i0 == 0) ? 1 : 0;
#pragma unroll
    for (int e = 0; e < NEXP; ++e)
      if (e != i0 && p[e] > p[i1]) i1 = e;
    float s = p[i0] + p[i1];
    float w0 = p[i0] / s, w1 = p[i1] / s;

    int pos0 = atomicAdd(&cnt[i0], 1);
    tlist[i0 * T_TOK + pos0] = t;
    twgt[i0 * T_TOK + pos0] = w0;
    int pos1 = atomicAdd(&cnt[i1], 1);
    tlist[i1 * T_TOK + pos1] = t;
    twgt[i1 * T_TOK + pos1] = w1;
  }
}

// ---------------------------------------------------------------------------
// Fused expert kernel. Block = 256 thr (8 waves), grid = (256 Mtiles, E).
// Per block: 32 gathered tokens (2 M-subtiles), all 1024 output columns.
// F-chunk = 64: GEMM1 (x@w1^T, x@w3^T) -> silu gate in LDS -> GEMM2 (hg@w2^T).
// Each wave computes BOTH M-subtiles so every B fragment feeds 2 WMMAs.
// WT = __bf16 (pre-converted weights) or float (fallback, cvt in flight).
// ---------------------------------------------------------------------------
template <typename WT>
__global__ __launch_bounds__(256) void moe_expert_kernel(
    const float* __restrict__ x, const WT* __restrict__ w1,
    const WT* __restrict__ w2, const WT* __restrict__ w3,
    const int* __restrict__ cnt, const int* __restrict__ tlist,
    const float* __restrict__ twgt, float* __restrict__ out) {
  const int e = blockIdx.y;
  const int mt = blockIdx.x;
  const int nTok = cnt[e];
  if (mt * 32 >= nTok) return;

  __shared__ __align__(16) __bf16 xlds[32 * HDIM];   // 64 KB
  __shared__ __align__(16) float g1[32 * 64];        // 8 KB
  __shared__ __align__(16) float g3[32 * 64];        // 8 KB
  __shared__ __align__(16) __bf16 hg[32 * 64];       // 4 KB
  __shared__ int   tokS[32];
  __shared__ float wS[32];

  const int tid = threadIdx.x;
  const int wave = tid >> 5;
  const int lane = tid & 31;
  const int col = lane & 15;
  const int hi = (lane >> 4) & 1;

  if (tid < 32) {
    int idx = mt * 32 + tid;
    if (idx < nTok) {
      tokS[tid] = tlist[e * T_TOK + idx];
      wS[tid] = twgt[e * T_TOK + idx];
    } else {
      tokS[tid] = 0;   // padded row: weight 0 => contributes exactly 0.0f
      wS[tid] = 0.f;
    }
  }
  __syncthreads();

  // Gather x tile -> LDS bf16. 8 threads per row, 128 cols each (float4).
  {
    const int r = tid >> 3;
    const int c0 = (tid & 7) * 128;
    const float* src = x + (size_t)tokS[r] * HDIM + c0;
    __bf16* dst = xlds + r * HDIM + c0;
#pragma unroll
    for (int i = 0; i < 32; ++i) {
      float4 f = ((const float4*)src)[i];
      dst[4 * i + 0] = (__bf16)f.x;
      dst[4 * i + 1] = (__bf16)f.y;
      dst[4 * i + 2] = (__bf16)f.z;
      dst[4 * i + 3] = (__bf16)f.w;
    }
  }
  __syncthreads();

  const WT* w1e = w1 + (size_t)e * FDIM * HDIM;
  const WT* w3e = w3 + (size_t)e * FDIM * HDIM;
  const WT* w2e = w2 + (size_t)e * HDIM * FDIM;

  v8f acc0[8], acc1[8];   // M-subtile 0 / 1, 8 N-tiles each (128 VGPRs)
#pragma unroll
  for (int nt = 0; nt < 8; ++nt)
#pragma unroll
    for (int v = 0; v < 8; ++v) { acc0[nt][v] = 0.f; acc1[nt][v] = 0.f; }

  const int mat = wave & 1;       // 0 -> w1 tile, 1 -> w3 tile
  const int fc = wave >> 1;       // which 16-wide f sub-tile of the 64 chunk
  const WT* wsrc = mat ? w3e : w1e;
  float* gdst = mat ? g3 : g1;

  for (int f0 = 0; f0 < FDIM; f0 += 64) {
    // ---- GEMM1: wave computes its (fc, mat) tile for BOTH M-subtiles,
    //      sharing each B fragment across the two WMMAs. K = HDIM.
    v8f c0, c1;
#pragma unroll
    for (int v = 0; v < 8; ++v) { c0[v] = 0.f; c1[v] = 0.f; }
    const WT* brow = wsrc + (size_t)(f0 + fc * 16 + col) * HDIM + hi * 16;
    if (f0 + 64 < FDIM)  // pull next F-chunk toward the caches
      __builtin_prefetch(wsrc + (size_t)(f0 + 64 + fc * 16 + col) * HDIM, 0, 1);
    for (int ks = 0; ks < HDIM / 32; ++ks) {
      v16bf b = ld_b(brow + ks * 32);
      v16bf a0 = lds_load_a(xlds, HDIM, lane, ks * 32);
      v16bf a1 = lds_load_a(xlds + 16 * HDIM, HDIM, lane, ks * 32);
      c0 = wmma_bf16(a0, b, c0);
      c1 = wmma_bf16(a1, b, c1);
    }
#pragma unroll
    for (int v = 0; v < 8; ++v) {
      gdst[(v + hi * 8) * 64 + fc * 16 + col] = c0[v];
      gdst[(16 + v + hi * 8) * 64 + fc * 16 + col] = c1[v];
    }
    __syncthreads();

    // ---- silu(g1) * g3 -> hg (bf16), 2048 elems / 256 threads
#pragma unroll
    for (int i = tid; i < 32 * 64; i += 256) {
      float a1v = g1[i];
      float s = a1v / (1.f + __expf(-a1v));
      hg[i] = (__bf16)(s * g3[i]);
    }
    __syncthreads();

    // ---- GEMM2: wave owns 128 output cols (8 N-tiles), K = 64 (2 steps);
    //      B fragment loaded once, used for both M-subtiles.
#pragma unroll
    for (int ks = 0; ks < 2; ++ks) {
      v16bf a20 = lds_load_a(hg, 64, lane, ks * 32);
      v16bf a21 = lds_load_a(hg + 16 * 64, 64, lane, ks * 32);
#pragma unroll
      for (int nt = 0; nt < 8; ++nt) {
        int h = wave * 128 + nt * 16 + col;
        v16bf b2 = ld_b(w2e + (size_t)h * FDIM + f0 + ks * 32 + hi * 16);
        acc0[nt] = wmma_bf16(a20, b2, acc0[nt]);
        acc1[nt] = wmma_bf16(a21, b2, acc1[nt]);
      }
    }
    __syncthreads();  // hg/g1/g3 reused next chunk
  }

  // ---- epilogue: scale rows by routing weight, accumulate into out.
  // Exactly two adds land on each output element (top-2) => deterministic.
#pragma unroll
  for (int nt = 0; nt < 8; ++nt) {
    int h = wave * 128 + nt * 16 + col;
#pragma unroll
    for (int v = 0; v < 8; ++v) {
      int r0 = v + hi * 8;
      int r1 = 16 + v + hi * 8;
      unsafeAtomicAdd(out + (size_t)tokS[r0] * HDIM + h, acc0[nt][v] * wS[r0]);
      unsafeAtomicAdd(out + (size_t)tokS[r1] * HDIM + h, acc1[nt][v] * wS[r1]);
    }
  }
}

// ---------------------------------------------------------------------------
extern "C" void kernel_launch(void* const* d_in, const int* in_sizes, int n_in,
                              void* d_out, int out_size, void* d_ws,
                              size_t ws_size, hipStream_t stream) {
  const float* x = (const float*)d_in[0];   // [B,S,H]
  const float* gw = (const float*)d_in[1];  // [E,H]
  const float* w1 = (const float*)d_in[2];  // [E,F,H]
  const float* w2 = (const float*)d_in[3];  // [E,H,F]
  const float* w3 = (const float*)d_in[4];  // [E,F,H]
  // d_in[5] = top_k (compile-time 2 here)

  float* out = (float*)d_out;                            // [T,H]
  float* logits = (float*)d_out + (size_t)T_TOK * HDIM;  // [T,E]

  char* ws = (char*)d_ws;
  int* cnt = (int*)ws;                                        // E ints
  int* tlist = (int*)(ws + 256);                              // E*T ints
  float* twgt = (float*)(ws + 256 + sizeof(int) * NEXP * T_TOK);

  hipMemsetAsync(cnt, 0, 256, stream);
  hipMemsetAsync(out, 0, sizeof(float) * (size_t)T_TOK * HDIM, stream);

  router_kernel<<<T_TOK / 8, 256, 0, stream>>>(x, gw, logits, cnt, tlist,
                                               twgt);

  const size_t WELEM = (size_t)NEXP * FDIM * HDIM;   // per weight tensor
  const size_t META = 1u << 20;                      // 1MB for routing data
  const size_t needBf16 = META + 3 * WELEM * sizeof(__bf16);

  dim3 grid(T_TOK / 32, NEXP, 1);  // over-provisioned; blocks early-exit

  if (ws_size >= needBf16) {
    // Pre-convert weights to bf16 once (bandwidth-bound), then bf16 GEMM.
    __bf16* w1b = (__bf16*)(ws + META);
    __bf16* w2b = w1b + WELEM;
    __bf16* w3b = w2b + WELEM;
    const int cvtBlocks = (int)(WELEM / 8 / 256);
    cvt_bf16_kernel<<<cvtBlocks, 256, 0, stream>>>(w1, w1b, WELEM);
    cvt_bf16_kernel<<<cvtBlocks, 256, 0, stream>>>(w2, w2b, WELEM);
    cvt_bf16_kernel<<<cvtBlocks, 256, 0, stream>>>(w3, w3b, WELEM);
    moe_expert_kernel<__bf16><<<grid, 256, 0, stream>>>(
        x, w1b, w2b, w3b, cnt, tlist, twgt, out);
  } else {
    // Fallback: convert weights in flight (more VALU per WMMA).
    moe_expert_kernel<float><<<grid, 256, 0, stream>>>(
        x, w1, w2, w3, cnt, tlist, twgt, out);
  }
}
